// MTGNNBlock_7593502179659
// MI455X (gfx1250) — matrix operand order repaired and checked by reading, not compiled, not run on hardware
//
#include <hip/hip_runtime.h>

// ---------------------------------------------------------------------------
// MTGNN block for MI455X (gfx1250). All matmuls via v_wmma_f32_16x16x32_bf16.
// B=16, C=32, N=512, T=256, K=2, DEPTH=2, ALPHA=0.05
//
// Layouts:
//   x (input, fixed):  [b][c][n][t]        f32
//   hp buffers:        [b*32+c][t][node]   bf16  (node-contiguous, MixProp)
//   hc buffers:        [b][n][t][c]        bf16  (channel-contiguous, MLP)
//   mo buffer:         [b][n][t][o]        bf16  (o-contiguous, final)
//   out (fixed):       [b][o][n][t]        f32
// ---------------------------------------------------------------------------

typedef __bf16 v16bf __attribute__((ext_vector_type(16)));
typedef float  v8f   __attribute__((ext_vector_type(8)));

#define NB 16
#define NC 32
#define NN 512
#define NT 256
#define ALPHA_F 0.05f

__device__ __forceinline__ v8f wmma_bf16(v16bf a, v16bf b, v8f c) {
  return __builtin_amdgcn_wmma_f32_16x16x32_bf16(
      false, a, false, b, (short)0, c, false, false);
}

__device__ __forceinline__ float fast_sigmoid(float x) {
  return 1.0f / (1.0f + __expf(-x));
}
__device__ __forceinline__ float fast_tanh(float x) {
  return 1.0f - 2.0f / (__expf(2.0f * x) + 1.0f);
}

// 16 contiguous bf16 at p (16B aligned): B-fragment (lane holds K=16*half+j
// when p points at row[16*half]).
__device__ __forceinline__ v16bf ld_frag16(const __bf16* p) {
  union { v16bf v; uint4 q[2]; } u;
  u.q[0] = *(const uint4*)(p);
  u.q[1] = *(const uint4*)(p + 8);
  return u.v;
}
// A-fragment: row of a [16 x 32] tile; lane half selects K run
// {half*16..+15}.
__device__ __forceinline__ v16bf ld_frag_a(const __bf16* row, int half) {
  union { v16bf v; uint4 q[2]; } u;
  u.q[0] = *(const uint4*)(row + half * 16);
  u.q[1] = *(const uint4*)(row + half * 16 + 8);
  return u.v;
}

union PackB8 { uint4 q; __bf16 e[8]; };

// ---------------------------------------------------------------------------
// 1) Row-normalize adjacency -> bf16
// ---------------------------------------------------------------------------
__global__ void __launch_bounds__(256) k_prep_adj(const float* __restrict__ adj,
                                                  __bf16* __restrict__ adjn) {
  int i = blockIdx.x, tid = threadIdx.x;
  __shared__ float red[256];
  float a0 = adj[i * NN + tid];
  float a1 = adj[i * NN + 256 + tid];
  red[tid] = a0 + a1;
  __syncthreads();
  for (int off = 128; off > 0; off >>= 1) {
    if (tid < off) red[tid] += red[tid + off];
    __syncthreads();
  }
  float inv = 1.0f / (red[0] + 1e-8f);
  adjn[i * NN + tid]       = (__bf16)(a0 * inv);
  adjn[i * NN + 256 + tid] = (__bf16)(a1 * inv);
}

// ---------------------------------------------------------------------------
// 2) Gated temporal conv -> h0 in both layouts.
//    Per (b,n): [32x32]x[32x256] x 2 taps x 2 branches. Tap1 reads x[t+1]
//    (zero pad col 256, per the reference's [..., -T:] crop).
// ---------------------------------------------------------------------------
__global__ void __launch_bounds__(256) k_tconv(const float* __restrict__ x,
                                               const float* __restrict__ fw,
                                               const float* __restrict__ fb,
                                               const float* __restrict__ gw,
                                               const float* __restrict__ gb,
                                               __bf16* __restrict__ h0p,
                                               __bf16* __restrict__ h0c) {
  int n = blockIdx.x, bb = blockIdx.y;
  __shared__ __attribute__((aligned(16))) __bf16 xsT[257][40]; // [t][c], row 256 = zeros
  __shared__ __attribute__((aligned(16))) __bf16 wf[2][NC][40];
  __shared__ __attribute__((aligned(16))) __bf16 wg[2][NC][40];
  __shared__ float bias[64];
  int tid = threadIdx.x;

  { // transpose-stage x row [32 x 256] f32 -> xsT[t][c] bf16
    int c = tid >> 3, t0 = (tid & 7) * 32;
    const float4* src =
        (const float4*)(x + ((((size_t)bb * NC + c) * NN + n) * NT + t0));
#pragma unroll
    for (int q = 0; q < 8; ++q) {
      float4 v = src[q];
      int t = t0 + q * 4;
      xsT[t + 0][c] = (__bf16)v.x; xsT[t + 1][c] = (__bf16)v.y;
      xsT[t + 2][c] = (__bf16)v.z; xsT[t + 3][c] = (__bf16)v.w;
    }
    if (tid < NC) xsT[256][tid] = (__bf16)0.0f;
  }
  { // weights -> bf16 LDS
    int e0 = tid * 8;
#pragma unroll
    for (int q = 0; q < 8; ++q) {
      int e = e0 + q;                 // e = (o*32+i)*2 + k
      int k = e & 1, oi = e >> 1, o = oi >> 5, i = oi & 31;
      wf[k][o][i] = (__bf16)fw[e];
      wg[k][o][i] = (__bf16)gw[e];
    }
    if (tid < NC) { bias[tid] = fb[tid]; bias[32 + tid] = gb[tid]; }
  }
  __syncthreads();

  int w = tid >> 5, lane = tid & 31, m = lane & 15, half = lane >> 4;

  v16bf af0[2], af1[2], ag0[2], ag1[2];
#pragma unroll
  for (int mt = 0; mt < 2; ++mt) {
    af0[mt] = ld_frag_a(&wf[0][mt * 16 + m][0], half);
    af1[mt] = ld_frag_a(&wf[1][mt * 16 + m][0], half);
    ag0[mt] = ld_frag_a(&wg[0][mt * 16 + m][0], half);
    ag1[mt] = ld_frag_a(&wg[1][mt * 16 + m][0], half);
  }

#pragma unroll
  for (int q = 0; q < 2; ++q) {
    int tt = w * 2 + q;
    int t = tt * 16 + m;
    v16bf b0 = ld_frag16(&xsT[t][half * 16]);       // tap0: x[t]
    v16bf b1 = ld_frag16(&xsT[t + 1][half * 16]);   // tap1: x[t+1]
#pragma unroll
    for (int mt = 0; mt < 2; ++mt) {
      v8f cf = {0.f,0.f,0.f,0.f,0.f,0.f,0.f,0.f};
      v8f cg = {0.f,0.f,0.f,0.f,0.f,0.f,0.f,0.f};
      cf = wmma_bf16(af0[mt], b0, cf);
      cf = wmma_bf16(af1[mt], b1, cf);
      cg = wmma_bf16(ag0[mt], b0, cg);
      cg = wmma_bf16(ag1[mt], b1, cg);
      PackB8 st;
#pragma unroll
      for (int r = 0; r < 8; ++r) {
        int o = mt * 16 + r + 8 * half;
        float f = fast_tanh(cf[r] + bias[o]);
        float g = fast_sigmoid(cg[r] + bias[32 + o]);
        float hv = f * g;
        st.e[r] = (__bf16)hv;
        // node-contiguous copy (scalar scatter over o)
        h0p[((size_t)(bb * NC + o) * NT + t) * NN + n] = (__bf16)hv;
      }
      // channel-contiguous copy (vector store: 8 consecutive o)
      *(uint4*)(h0c + (((size_t)bb * NN + n) * NT + t) * NC + mt * 16 + 8 * half) = st.q;
    }
  }
}

// ---------------------------------------------------------------------------
// 3) MixProp hop: hout = ALPHA*h0 + (1-ALPHA)*(A @ hin) over nodes, per (b,c).
//    Software-pipelined: double-buffered LDS panels, next K-step's global
//    loads issued before the current step's WMMA chain; one barrier/step.
// ---------------------------------------------------------------------------
__global__ void __launch_bounds__(256) k_mixprop(const __bf16* __restrict__ adjn,
                                                 const __bf16* __restrict__ hin_p,
                                                 const __bf16* __restrict__ h0_p,
                                                 __bf16* __restrict__ hout_p,
                                                 __bf16* __restrict__ hout_c,
                                                 int write_primary) {
  int mb = blockIdx.x * 64;      // node tile base (M)
  int bc = blockIdx.y;           // b*32 + c
  int bb = bc >> 5, cc = bc & 31;
  __shared__ __attribute__((aligned(16))) __bf16 At[2][64][40];
  __shared__ __attribute__((aligned(16))) __bf16 BtT[2][256][40]; // [t][k]
  int tid = threadIdx.x, w = tid >> 5, lane = tid & 31;
  int half = lane >> 4, m = lane & 15;
  int mw = w & 3;
  int tb = (w >> 2) * 8;

  v8f acc[8];
#pragma unroll
  for (int i = 0; i < 8; ++i) acc[i] = (v8f){0.f,0.f,0.f,0.f,0.f,0.f,0.f,0.f};

  // per-thread staging addresses
  int mi = tid >> 2, seg = (tid & 3) * 8;
  const __bf16* aptr = adjn + (size_t)(mb + mi) * NN + seg;       // + kk
  const __bf16* bptr = hin_p + (size_t)bc * NT * NN + (size_t)tid * NN; // + kk

  { // prologue: stage K-step 0 into buffer 0
    *(uint4*)&At[0][mi][seg] = *(const uint4*)(aptr);
#pragma unroll
    for (int q = 0; q < 4; ++q)
      *(uint4*)&BtT[0][tid][q * 8] = *(const uint4*)(bptr + q * 8);
  }
  __syncthreads();

  const int NSTEP = NN / 32;  // 16
#pragma unroll 2
  for (int s = 0; s < NSTEP; ++s) {
    int cur = s & 1, nxt = cur ^ 1;
    uint4 ra, rb[4];
    if (s + 1 < NSTEP) {       // issue next step's global loads early
      int kk = (s + 1) * 32;
      ra = *(const uint4*)(aptr + kk);
#pragma unroll
      for (int q = 0; q < 4; ++q) rb[q] = *(const uint4*)(bptr + kk + q * 8);
    }
    v16bf a = ld_frag_a(&At[cur][mw * 16 + m][0], half);
#pragma unroll
    for (int q = 0; q < 8; ++q) {
      int t = (tb + q) * 16 + m;
      v16bf bfrag = ld_frag16(&BtT[cur][t][half * 16]);
      acc[q] = wmma_bf16(a, bfrag, acc[q]);
    }
    if (s + 1 < NSTEP) {       // park loads into the alternate panel
      *(uint4*)&At[nxt][mi][seg] = ra;
#pragma unroll
      for (int q = 0; q < 4; ++q) *(uint4*)&BtT[nxt][tid][q * 8] = rb[q];
    }
    __syncthreads();
  }

  const __bf16* h0_bc = h0_p + (size_t)bc * NT * NN;
  __bf16* hop_bc = hout_p + (size_t)bc * NT * NN;
  int i0 = mb + mw * 16 + 8 * half;   // 8 consecutive nodes per lane
#pragma unroll
  for (int q = 0; q < 8; ++q) {
    int t = (tb + q) * 16 + m;
    PackB8 h0u, st;
    h0u.q = *(const uint4*)(h0_bc + (size_t)t * NN + i0);
#pragma unroll
    for (int r = 0; r < 8; ++r) {
      float v = ALPHA_F * (float)h0u.e[r] + (1.0f - ALPHA_F) * acc[q][r];
      st.e[r] = (__bf16)v;
      // channel-contiguous copy (scalar scatter over node)
      hout_c[(((size_t)bb * NN + (i0 + r)) * NT + t) * NC + cc] = st.e[r];
    }
    if (write_primary)
      *(uint4*)(hop_bc + (size_t)t * NN + i0) = st.q;
  }
}

// ---------------------------------------------------------------------------
// 4) MLP mix [32x96] over channel-contiguous {h0c,h1c,h2c}; B-fragments come
//    straight from global (2x b128 per slab). Accumulates BN stats.
// ---------------------------------------------------------------------------
__global__ void __launch_bounds__(256) k_mlp(const __bf16* __restrict__ h0c,
                                             const __bf16* __restrict__ h1c,
                                             const __bf16* __restrict__ h2c,
                                             const float* __restrict__ mlp_w,
                                             const float* __restrict__ mlp_b,
                                             __bf16* __restrict__ mo,
                                             float* __restrict__ stats) {
  int n = blockIdx.x, bb = blockIdx.y;
  __shared__ __attribute__((aligned(16))) __bf16 wsw[NC][104];
  __shared__ float ssum[NC], ssq[NC], biasl[NC];
  int tid = threadIdx.x;

  for (int e = tid; e < NC * 96; e += 256) wsw[e / 96][e % 96] = (__bf16)mlp_w[e];
  if (tid < NC) { biasl[tid] = mlp_b[tid]; ssum[tid] = 0.f; ssq[tid] = 0.f; }
  __syncthreads();

  int w = tid >> 5, lane = tid & 31, half = lane >> 4, m = lane & 15;
  v16bf a[2][3];
#pragma unroll
  for (int mt = 0; mt < 2; ++mt)
#pragma unroll
    for (int s = 0; s < 3; ++s)
      a[mt][s] = ld_frag_a(&wsw[mt * 16 + m][s * 32], half);

  size_t rowbase = ((size_t)bb * NN + n) * NT;  // * NC applied below
#pragma unroll
  for (int q = 0; q < 2; ++q) {
    int t = (w * 2 + q) * 16 + m;
    size_t off = (rowbase + t) * NC + 16 * half;
    v16bf bf0 = ld_frag16(h0c + off);
    v16bf bf1 = ld_frag16(h1c + off);
    v16bf bf2 = ld_frag16(h2c + off);
#pragma unroll
    for (int mt = 0; mt < 2; ++mt) {
      v8f c = {0.f,0.f,0.f,0.f,0.f,0.f,0.f,0.f};
      c = wmma_bf16(a[mt][0], bf0, c);
      c = wmma_bf16(a[mt][1], bf1, c);
      c = wmma_bf16(a[mt][2], bf2, c);
      PackB8 st;
#pragma unroll
      for (int r = 0; r < 8; ++r) {
        int o = mt * 16 + r + 8 * half;
        float v = c[r] + biasl[o];
        st.e[r] = (__bf16)v;
        atomicAdd(&ssum[o], v);
        atomicAdd(&ssq[o], v * v);
      }
      *(uint4*)(mo + (rowbase + t) * NC + mt * 16 + 8 * half) = st.q;
    }
  }
  __syncthreads();
  if (tid < NC) {
    atomicAdd(&stats[tid], ssum[tid]);
    atomicAdd(&stats[NC + tid], ssq[tid]);
  }
}

// ---------------------------------------------------------------------------
// 5) Fold BN stats into per-channel scale/shift.
// ---------------------------------------------------------------------------
__global__ void k_bnpar(const float* __restrict__ stats,
                        const float* __restrict__ gamma,
                        const float* __restrict__ beta,
                        float* __restrict__ bnpar) {
  int c = threadIdx.x;
  float cnt = (float)NB * (float)NN * (float)NT;
  float mean = stats[c] / cnt;
  float var  = stats[NC + c] / cnt - mean * mean;
  float istd = rsqrtf(var + 1e-5f);
  float sc = gamma[c] * istd;
  bnpar[c]      = sc;
  bnpar[NC + c] = beta[c] - mean * sc;
}

// ---------------------------------------------------------------------------
// 6) Residual [32x32] WMMA on bf16 x + BN apply + ReLU -> f32 out [b][o][n][t].
// ---------------------------------------------------------------------------
__global__ void __launch_bounds__(256) k_final(const float* __restrict__ x,
                                               const float* __restrict__ res_w,
                                               const float* __restrict__ res_b,
                                               const __bf16* __restrict__ mo,
                                               const float* __restrict__ bnpar,
                                               float* __restrict__ out) {
  int n = blockIdx.x, bb = blockIdx.y;
  __shared__ __attribute__((aligned(16))) __bf16 xsT[NT][40]; // [t][c]
  __shared__ __attribute__((aligned(16))) __bf16 wr[NC][40];
  __shared__ float sc[NC], sh[NC], rb[NC];
  int tid = threadIdx.x;
  {
    int c = tid >> 3, t0 = (tid & 7) * 32;
    const float4* src =
        (const float4*)(x + ((((size_t)bb * NC + c) * NN + n) * NT + t0));
#pragma unroll
    for (int q = 0; q < 8; ++q) {
      float4 v = src[q];
      int t = t0 + q * 4;
      xsT[t + 0][c] = (__bf16)v.x; xsT[t + 1][c] = (__bf16)v.y;
      xsT[t + 2][c] = (__bf16)v.z; xsT[t + 3][c] = (__bf16)v.w;
    }
  }
  for (int e = tid; e < NC * NC; e += 256) wr[e >> 5][e & 31] = (__bf16)res_w[e];
  if (tid < NC) { sc[tid] = bnpar[tid]; sh[tid] = bnpar[NC + tid]; rb[tid] = res_b[tid]; }
  __syncthreads();

  int w = tid >> 5, lane = tid & 31, half = lane >> 4, m = lane & 15;
  v16bf a[2];
#pragma unroll
  for (int mt = 0; mt < 2; ++mt) a[mt] = ld_frag_a(&wr[mt * 16 + m][0], half);

#pragma unroll
  for (int q = 0; q < 2; ++q) {
    int t = (w * 2 + q) * 16 + m;
    v16bf bfrag = ld_frag16(&xsT[t][half * 16]);
#pragma unroll
    for (int mt = 0; mt < 2; ++mt) {
      v8f c = {0.f,0.f,0.f,0.f,0.f,0.f,0.f,0.f};
      c = wmma_bf16(a[mt], bfrag, c);
      PackB8 mu;
      mu.q = *(const uint4*)(mo + (((size_t)bb * NN + n) * NT + t) * NC +
                             mt * 16 + 8 * half);
#pragma unroll
      for (int r = 0; r < 8; ++r) {
        int o = mt * 16 + r + 8 * half;
        float v = sc[o] * (float)mu.e[r] + sh[o] + c[r] + rb[o];
        out[(((size_t)bb * NC + o) * NN + n) * NT + t] = v > 0.f ? v : 0.f;
      }
    }
  }
}

// ---------------------------------------------------------------------------
// Workspace (bytes):
//   [0, 512K)            adj_n bf16
//   [512K, +256)         stats (sum[32], sumsq[32])
//   [512K+256, +256)     bnpar (scale[32], shift[32])
//   [1M ...)             h0p, h0c, h1p, h1c, h2c, mo : 6 x 128 MiB bf16
// ---------------------------------------------------------------------------
extern "C" void kernel_launch(void* const* d_in, const int* in_sizes, int n_in,
                              void* d_out, int out_size, void* d_ws, size_t ws_size,
                              hipStream_t stream) {
  const float* x      = (const float*)d_in[0];
  const float* adj    = (const float*)d_in[1];
  const float* filt_w = (const float*)d_in[2];
  const float* filt_b = (const float*)d_in[3];
  const float* gate_w = (const float*)d_in[4];
  const float* gate_b = (const float*)d_in[5];
  const float* mlp_w  = (const float*)d_in[6];
  const float* mlp_b  = (const float*)d_in[7];
  const float* res_w  = (const float*)d_in[8];
  const float* res_b  = (const float*)d_in[9];
  const float* gamma  = (const float*)d_in[10];
  const float* beta   = (const float*)d_in[11];

  char* wsb = (char*)d_ws;
  const size_t ELEMS = (size_t)NB * NC * NN * NT; // 67,108,864
  __bf16* adjn  = (__bf16*)wsb;
  float*  stats = (float*)(wsb + 524288);
  float*  bnpar = stats + 64;
  __bf16* h0p = (__bf16*)(wsb + (1 << 20));
  __bf16* h0c = h0p + ELEMS;
  __bf16* h1p = h0c + ELEMS;
  __bf16* h1c = h1p + ELEMS;
  __bf16* h2c = h1c + ELEMS;
  __bf16* mo  = h2c + ELEMS;
  float* out = (float*)d_out;

  hipMemsetAsync(stats, 0, 64 * sizeof(float), stream);
  k_prep_adj<<<NN, 256, 0, stream>>>(adj, adjn);
  k_tconv<<<dim3(NN, NB), 256, 0, stream>>>(x, filt_w, filt_b, gate_w, gate_b,
                                            h0p, h0c);
  k_mixprop<<<dim3(NN / 64, NB * NC), 256, 0, stream>>>(adjn, h0p, h0p,
                                                        h1p, h1c, 1);
  k_mixprop<<<dim3(NN / 64, NB * NC), 256, 0, stream>>>(adjn, h1p, h0p,
                                                        h1p /*unused*/, h2c, 0);
  k_mlp<<<dim3(NN, NB), 256, 0, stream>>>(h0c, h1c, h2c, mlp_w, mlp_b, mo, stats);
  k_bnpar<<<1, NC, 0, stream>>>(stats, gamma, beta, bnpar);
  k_final<<<dim3(NN, NB), 256, 0, stream>>>(x, res_w, res_b, mo, bnpar, out);
}
